// Decoder_91122026151952
// MI455X (gfx1250) — compile-verified, matrix operand order checked by reading, not run on hardware
//
#include <hip/hip_runtime.h>

typedef __attribute__((ext_vector_type(16))) _Float16 v16h;
typedef __attribute__((ext_vector_type(8)))  _Float16 v8h;
typedef __attribute__((ext_vector_type(8)))  float    v8f;

#define B_    1024
#define T_    240
#define H_    180
#define OUT_  40
#define KP    192            // K padded to 6*32 for gemm1
#define NT1   12             // 16-wide hcol tiles per gate block (192/16)
#define KS1   6              // k-steps (32 each) for gemm1
#define TCH   15             // t-chunks of 16 (240/16)
#define BB    16             // batches per workgroup
#define TT    16             // timesteps per workgroup
#define KTOT  (T_*H_)        // 43200
#define KS2TOT (KTOT/32)     // 1350 global k-steps for gemm2
#define KS2WG  (TT*H_/32)    // 90 k-steps per workgroup

// workspace layout
#define BPACK_BYTES  (3*NT1*KS1*32*16*2)        // 221184   (W_ih i/g/o fragments, f16)
#define WOPACK_BYTES (3*(size_t)KS2TOT*32*16*2) // 4147200  (W_out fragments, f16)
#define WS_BPACK   0
#define WS_WOPACK  (WS_BPACK + BPACK_BYTES)
#define WS_PARTIAL (WS_WOPACK + WOPACK_BYTES)

#define LOG2E 1.4426950408889634f

// Branch-free activations on CDNA5 TRANS units (v_exp_f32 is exp2; v_rcp_f32).
// These co-execute with WMMA (XDL ops are tracked as TRANS for scheduling).
__device__ __forceinline__ float fast_sigmoid(float x) {
    // 1/(1+e^-x); exp2->inf for very negative x gives rcp(inf)=0: graceful.
    return __builtin_amdgcn_rcpf(1.f + __builtin_amdgcn_exp2f(-LOG2E * x));
}
__device__ __forceinline__ float fast_tanh(float x) {
    // tanh(x) = 1 - 2/(e^{2x}+1); exp2->inf gives 1-0=1: graceful, no NaN.
    return 1.f - 2.f * __builtin_amdgcn_rcpf(1.f + __builtin_amdgcn_exp2f(2.f * LOG2E * x));
}

// ---------------------------------------------------------------------------
// Pack W_ih (only i/g/o blocks) into WMMA B-fragment layout, f16, zero padded.
// B 32x16 (KxN): lane = (k_half>=1)*16 + n, half j -> K = base + (j<8 ? j : j+8)
// linear index = ((g*NT1 + nt)*KS1 + ks)*512 + lane*16 + h
// ---------------------------------------------------------------------------
__global__ void k_pack_wih(const float* __restrict__ W_ih, _Float16* __restrict__ Bpack) {
    int idx = blockIdx.x * 256 + threadIdx.x;          // < 110592
    int h    = idx & 15;
    int lane = (idx >> 4) & 31;
    int rest = idx >> 9;
    int ks = rest % KS1; rest /= KS1;
    int nt = rest % NT1;
    int g  = rest / NT1;
    int n    = nt * 16 + (lane & 15);
    int koff = ((lane >> 4) << 3) + ((h < 8) ? h : h + 8);
    int k    = ks * 32 + koff;
    int grow = (g == 0) ? 0 : (g == 1 ? 2 * H_ : 3 * H_);  // i, g, o gate rows
    float v = 0.f;
    if (n < H_ && k < H_) v = W_ih[(size_t)(grow + n) * H_ + k];
    Bpack[idx] = (_Float16)v;
}

// ---------------------------------------------------------------------------
// Pack W_out [40 x 43200] into B-fragment layout, f16 (N padded 40 -> 48).
// linear index = ((nt2*KS2TOT + ks)*32 + lane)*16 + h
// ---------------------------------------------------------------------------
__global__ void k_pack_wout(const float* __restrict__ W_out, _Float16* __restrict__ Wopack) {
    int idx = blockIdx.x * 256 + threadIdx.x;          // < 2073600
    int h    = idx & 15;
    int lane = (idx >> 4) & 31;
    int rest = idx >> 9;
    int ks  = rest % KS2TOT;
    int nt2 = rest / KS2TOT;
    int n    = nt2 * 16 + (lane & 15);
    int koff = ((lane >> 4) << 3) + ((h < 8) ? h : h + 8);
    int k    = ks * 32 + koff;
    float v = 0.f;
    if (n < OUT_) v = W_out[(size_t)n * KTOT + k];
    Wopack[idx] = (_Float16)v;
}

// ---------------------------------------------------------------------------
// Fused kernel: one workgroup = 16 batches x 16 timesteps.
//   Stage 1: x tile -> LDS (f16, K padded to 192)
//   Stage 2: GEMM1 via WMMA (i/g/o gates in 3 register accumulators seeded
//            with the bias), branch-free sigmoid/tanh, h tile -> LDS (f16)
//   Stage 3: GEMM2 split-K partial (h[16b x 2880] x W_outT), LDS reduction,
//            partial [16,40] -> workspace.
// 12 waves (384 threads): each wave owns one 16-col hcol tile in GEMM1.
// ---------------------------------------------------------------------------
__global__ __launch_bounds__(384, 1)
void k_fused(const float* __restrict__ x,
             const _Float16* __restrict__ Bpack,
             const float* __restrict__ b_ih,
             const float* __restrict__ b_hh,
             const _Float16* __restrict__ Wopack,
             float* __restrict__ partial) {
    extern __shared__ char smem[];
    _Float16* sA   = (_Float16*)smem;                               // 256*192 f16 = 96 KB
    _Float16* sH   = (_Float16*)(smem + 256 * KP * 2);              // 16*2880 f16 = 90 KB
    float*    sRed = (float*)(smem + 256 * KP * 2 + BB * TT * H_ * 2); // 12*3*32*8 f32 = 36 KB

    int wg  = blockIdx.x;
    int tch = wg % TCH;
    int b0  = (wg / TCH) * BB;
    int t0  = tch * TT;
    int tid = threadIdx.x;
    int wave = tid >> 5, lane = tid & 31;
    int kb = (lane >> 4) << 3;   // K-base (0 or 8) for A/B fragment chunks

    // ---- Stage 1: load x tile, convert f32->f16, pad K to 192. row r = t*16 + b
    for (int r = wave; r < BB * TT; r += 12) {
        int bl = r & 15, tl = r >> 4;
        const float* src = x + ((size_t)(b0 + bl) * T_ + (t0 + tl)) * H_;
        for (int c = lane; c < KP; c += 32)
            sA[r * KP + c] = (_Float16)((c < H_) ? src[c] : 0.f);
    }
    __syncthreads();

    // ---- Stage 2: GEMM1. This wave owns hcol tile `nt`.
    int nt = wave;  // 12 waves == NT1 tiles, perfectly balanced
    const v16h* bp = (const v16h*)Bpack;
    v16h bfr[3][KS1];
#pragma unroll
    for (int g = 0; g < 3; ++g)
#pragma unroll
        for (int ks = 0; ks < KS1; ++ks)
            bfr[g][ks] = bp[((g * NT1 + nt) * KS1 + ks) * 32 + lane];

    int col = nt * 16 + (lane & 15);
    float bias0 = 0.f, bias1 = 0.f, bias2 = 0.f;
    if (col < H_) {
        bias0 = b_ih[col]           + b_hh[col];            // i gate
        bias1 = b_ih[2 * H_ + col]  + b_hh[2 * H_ + col];   // g gate
        bias2 = b_ih[3 * H_ + col]  + b_hh[3 * H_ + col];   // o gate
    }

    for (int rt = 0; rt < TT; ++rt) {
        // Seed accumulators with the bias: D = A*B + C does the add for free.
        v8f a0, a1, a2;
#pragma unroll
        for (int j = 0; j < 8; ++j) { a0[j] = bias0; a1[j] = bias1; a2[j] = bias2; }
        const _Float16* arow = sA + (rt * 16 + (lane & 15)) * KP + kb;
#pragma unroll
        for (int ks = 0; ks < KS1; ++ks) {
            v8h lo = *(const v8h*)(arow + ks * 32);
            v8h hi = *(const v8h*)(arow + ks * 32 + 16);
            v16h a = __builtin_shufflevector(lo, hi,
                0, 1, 2, 3, 4, 5, 6, 7, 8, 9, 10, 11, 12, 13, 14, 15);
            a0 = __builtin_amdgcn_wmma_f32_16x16x32_f16(false, a, false, bfr[0][ks], (short)0, a0, false, false);
            a1 = __builtin_amdgcn_wmma_f32_16x16x32_f16(false, a, false, bfr[1][ks], (short)0, a1, false, false);
            a2 = __builtin_amdgcn_wmma_f32_16x16x32_f16(false, a, false, bfr[2][ks], (short)0, a2, false, false);
        }
        if (col < H_) {
            int blb = (lane >> 4) << 3;  // C layout: M = j + 8*(lane>=16)
#pragma unroll
            for (int j = 0; j < 8; ++j) {
                float cc = fast_sigmoid(a0[j]) * fast_tanh(a1[j]);
                float hh = fast_sigmoid(a2[j]) * fast_tanh(cc);
                sH[(size_t)(j + blb) * (TT * H_) + rt * H_ + col] = (_Float16)hh;
            }
        }
    }
    __syncthreads();

    // ---- Stage 3: GEMM2 partial. A = h[16b x 2880] (LDS), B = W_out frags (L2).
    v8f o0 = {}, o1 = {}, o2 = {};
    const v16h* wp = (const v16h*)Wopack;
    for (int ks = wave; ks < KS2WG; ks += 12) {
        const _Float16* hrow = sH + (lane & 15) * (TT * H_) + ks * 32 + kb;
        v8h lo = *(const v8h*)hrow;
        v8h hi = *(const v8h*)(hrow + 16);
        v16h a = __builtin_shufflevector(lo, hi,
            0, 1, 2, 3, 4, 5, 6, 7, 8, 9, 10, 11, 12, 13, 14, 15);
        size_t ksg = (size_t)tch * KS2WG + ks;
        v16h w0 = wp[(0 * (size_t)KS2TOT + ksg) * 32 + lane];
        v16h w1 = wp[(1 * (size_t)KS2TOT + ksg) * 32 + lane];
        v16h w2 = wp[(2 * (size_t)KS2TOT + ksg) * 32 + lane];
        o0 = __builtin_amdgcn_wmma_f32_16x16x32_f16(false, a, false, w0, (short)0, o0, false, false);
        o1 = __builtin_amdgcn_wmma_f32_16x16x32_f16(false, a, false, w1, (short)0, o1, false, false);
        o2 = __builtin_amdgcn_wmma_f32_16x16x32_f16(false, a, false, w2, (short)0, o2, false, false);
    }
    *(v8f*)(sRed + (((wave * 3 + 0) * 32) + lane) * 8) = o0;
    *(v8f*)(sRed + (((wave * 3 + 1) * 32) + lane) * 8) = o1;
    *(v8f*)(sRed + (((wave * 3 + 2) * 32) + lane) * 8) = o2;
    __syncthreads();

    // Deterministic reduction over the 12 waves, write split-K partial.
    for (int idx = tid; idx < 3 * 32 * 8; idx += 384) {
        int nt2   = idx >> 8;
        int rem   = idx & 255;
        int lane2 = rem >> 3, j = rem & 7;
        float s = 0.f;
#pragma unroll
        for (int w = 0; w < 12; ++w)
            s += sRed[(((w * 3 + nt2) * 32) + lane2) * 8 + j];
        int n  = nt2 * 16 + (lane2 & 15);
        int bl = j + ((lane2 >> 4) << 3);
        if (n < OUT_)
            partial[((size_t)tch * B_ + (b0 + bl)) * OUT_ + n] = s;
    }
}

// ---------------------------------------------------------------------------
// Reduce the 15 split-K partials, add bias, softmax over groups of 10.
// ---------------------------------------------------------------------------
__global__ void k_out(const float* __restrict__ partial,
                      const float* __restrict__ b_out,
                      float* __restrict__ out) {
    int b = blockIdx.x;
    int t = threadIdx.x;
    __shared__ float s[OUT_];
    if (t < OUT_) {
        float v = b_out[t];
        for (int c = 0; c < TCH; ++c)
            v += partial[((size_t)c * B_ + b) * OUT_ + t];
        s[t] = v;
    }
    __syncthreads();
    if (t < 4) {
        float m = -1e30f;
        for (int j = 0; j < 10; ++j) m = fmaxf(m, s[t * 10 + j]);
        float e[10], sum = 0.f;
        for (int j = 0; j < 10; ++j) {
            e[j] = __builtin_amdgcn_exp2f(LOG2E * (s[t * 10 + j] - m));
            sum += e[j];
        }
        float inv = 1.f / sum;
        for (int j = 0; j < 10; ++j)
            out[(size_t)b * OUT_ + t * 10 + j] = e[j] * inv;
    }
}

extern "C" void kernel_launch(void* const* d_in, const int* in_sizes, int n_in,
                              void* d_out, int out_size, void* d_ws, size_t ws_size,
                              hipStream_t stream) {
    (void)in_sizes; (void)n_in; (void)out_size; (void)ws_size;
    const float* x     = (const float*)d_in[0];
    const float* W_ih  = (const float*)d_in[1];
    // d_in[2] = W_hh: unused (zero initial hidden state -> term vanishes)
    const float* b_ih  = (const float*)d_in[3];
    const float* b_hh  = (const float*)d_in[4];
    const float* W_out = (const float*)d_in[5];
    const float* b_out = (const float*)d_in[6];

    char* ws = (char*)d_ws;
    _Float16* Bpack   = (_Float16*)(ws + WS_BPACK);
    _Float16* Wopack  = (_Float16*)(ws + WS_WOPACK);
    float*    partial = (float*)(ws + WS_PARTIAL);
    float*    out     = (float*)d_out;

    k_pack_wih <<<(BPACK_BYTES / 2) / 256, 256, 0, stream>>>(W_ih, Bpack);
    k_pack_wout<<<(int)((WOPACK_BYTES / 2) / 256), 256, 0, stream>>>(W_out, Wopack);

    size_t smem = (size_t)256 * KP * 2 + (size_t)BB * TT * H_ * 2 + (size_t)12 * 3 * 32 * 8 * 4;
    k_fused<<<(B_ / BB) * TCH, 384, smem, stream>>>(x, Bpack, b_ih, b_hh, Wopack, partial);

    k_out<<<B_, 64, 0, stream>>>(partial, b_out, out);
}
